// BasicLayer_81827716924001
// MI455X (gfx1250) — matrix operand order repaired and checked by reading, not compiled, not run on hardware
//
#include <hip/hip_runtime.h>
#include <hip/hip_bf16.h>
#include <math.h>

#define NROWS 160000
#define DIMC  128
#define HIDC  512
#define KK    49
#define DEPTHL 2

typedef __attribute__((ext_vector_type(16))) __bf16 v16bf;
typedef __attribute__((ext_vector_type(8)))  float  v8f;

union AV { v16bf v; uint4 q[2]; unsigned int u[8]; };

__device__ __forceinline__ unsigned int packbf(float a, float b) {
    union { float f; unsigned int u; } ua, ub;
    ua.f = a; ub.f = b;
    unsigned int x = ua.u, y = ub.u;
    unsigned int lo = (x + 0x7FFFu + ((x >> 16) & 1u)) >> 16;  // RNE f32->bf16
    unsigned int hi = (y + 0x7FFFu + ((y >> 16) & 1u)) >> 16;
    return (lo & 0xFFFFu) | (hi << 16);
}

// Swizzled B layout: plane p in {0,1} of PLANE dwords; group g' = (ks*N + col)*2 + half
// holds 4 dwords (c=0..3) at offset p*PLANE + g'*4 + c, carrying bf16 K-pair
// k2 = ks*16 + p*8 + half*4 + c at column col. Lane fetch = 2x ds_load_b128.

// ---------------------------------------------------------------------------
// K0: swizzle-pack W1 (bf16 pairs), cb[i][n] = b2 + grn_b @ w2, zero sumsq.
// ---------------------------------------------------------------------------
__global__ void k_prep(const float* __restrict__ w1, const float* __restrict__ grn_b,
                       const float* __restrict__ w2, const float* __restrict__ b2,
                       unsigned int* __restrict__ w1p, float* __restrict__ cb,
                       float* __restrict__ sumsq) {
    int t = blockIdx.x * blockDim.x + threadIdx.x;
    if (t < DEPTHL * 32768) {                    // W1: per layer 2 planes x 16384 dwords
        int i  = t >> 15;
        int r  = t & 32767;
        int p  = r >> 14;
        int rr = r & 16383;
        int g  = rr >> 2, c = rr & 3;
        int half = g & 1, rest = g >> 1;
        int col = rest & 511, ks = rest >> 9;
        int k2  = ks * 16 + p * 8 + half * 4 + c;
        const float* w = w1 + i * DIMC * HIDC;
        w1p[t] = packbf(w[(2 * k2) * HIDC + col], w[(2 * k2 + 1) * HIDC + col]);
    }
    if (t < DEPTHL * DIMC) {
        int i = t / DIMC, n = t - i * DIMC;
        float s = b2[i * DIMC + n];
        const float* w = w2 + i * HIDC * DIMC;
        const float* g = grn_b + i * HIDC;
        for (int k = 0; k < HIDC; ++k) s += g[k] * w[k * DIMC + n];
        cb[t] = s;
    }
    if (t < DEPTHL * HIDC) sumsq[t] = 0.0f;
}

// ---------------------------------------------------------------------------
// K1: dwconv(gather) + LayerNorm + GEMM1(bf16 WMMA, dual accumulators) + GELU.
// ---------------------------------------------------------------------------
__device__ __forceinline__ void epilogue1(v8f acc, int nt, int blockRow, int mbase,
                                          int half, int ccol, const float* b1s,
                                          unsigned int* Hp, float* sumsq) {
    const int col  = nt * 16 + ccol;
    const float bias = b1s[col];
    float ss = 0.f;
#pragma unroll
    for (int r = 0; r < 8; ++r) {
        float v  = acc[r] + bias;
        float gv = 0.5f * v * (1.0f + erff(v * 0.7071067811865476f));
        ss += gv * gv;
        float other = __shfl_xor(gv, 1);
        if (!(ccol & 1)) {
            int row = blockRow + mbase + r + 8 * half;
            Hp[(long)row * 256 + nt * 8 + (ccol >> 1)] = packbf(gv, other);
        }
    }
    ss += __shfl_xor(ss, 16);
    if (half == 0) atomicAdd(&sumsq[col], ss);
}

__global__ __launch_bounds__(256) void k_layer_front(
        const float* __restrict__ xin, const int* __restrict__ nbr,
        const float* __restrict__ wdw_g, const float* __restrict__ bdw_g,
        const float* __restrict__ lng_g, const float* __restrict__ lnb_g,
        const unsigned int* __restrict__ w1p_g, const float* __restrict__ b1_g,
        unsigned int* __restrict__ Hp, float* __restrict__ sumsq) {
    __shared__ unsigned int Bsw[32768];     // W1 swizzled (2 planes); aliases w_dw f32 first
    __shared__ unsigned int Ap[128 * 68];   // A tile bf16 pairs, padded row stride
    __shared__ float bdw[DIMC], lng[DIMC], lnb[DIMC], b1s[HIDC];

    const int tid  = threadIdx.x;
    const int lane = tid & 31, wv = tid >> 5;
    const int blockRow = blockIdx.x * 128;

    float* wdw = (float*)Bsw;
    for (int t = tid; t < KK * DIMC; t += 256) wdw[t] = wdw_g[t];
    for (int t = tid; t < DIMC; t += 256) { bdw[t] = bdw_g[t]; lng[t] = lng_g[t]; lnb[t] = lnb_g[t]; }
    for (int t = tid; t < HIDC; t += 256) b1s[t] = b1_g[t];
    __syncthreads();

    // ---- depthwise conv + LN: wave wv owns rows wv*16..wv*16+15 (4 ch/lane) ----
    for (int rr = 0; rr < 16; ++rr) {
        const int m   = wv * 16 + rr;
        const int row = blockRow + m;
        float a0 = 0.f, a1 = 0.f, a2 = 0.f, a3 = 0.f;
        const int* nb = nbr + (long)row * KK;
        for (int k = 0; k < KK; ++k) {
            int idx = nb[k];
            if ((unsigned)idx < (unsigned)NROWS) {
                const float4 xv = *(const float4*)(xin + (long)idx * DIMC + lane * 4);
                const float4 wc = *(const float4*)(wdw + k * DIMC + lane * 4);
                a0 += xv.x * wc.x; a1 += xv.y * wc.y; a2 += xv.z * wc.z; a3 += xv.w * wc.w;
            }
        }
        const float4 bv = *(const float4*)(bdw + lane * 4);
        a0 += bv.x; a1 += bv.y; a2 += bv.z; a3 += bv.w;
        float s = a0 + a1 + a2 + a3;
        for (int o = 16; o >= 1; o >>= 1) s += __shfl_xor(s, o);
        const float mu = s * (1.0f / 128.0f);
        float d0 = a0 - mu, d1 = a1 - mu, d2 = a2 - mu, d3 = a3 - mu;
        float q = d0 * d0 + d1 * d1 + d2 * d2 + d3 * d3;
        for (int o = 16; o >= 1; o >>= 1) q += __shfl_xor(q, o);
        const float rstd = rsqrtf(q * (1.0f / 128.0f) + 1e-6f);
        const float4 gv = *(const float4*)(lng + lane * 4);
        const float4 bb = *(const float4*)(lnb + lane * 4);
        const float n0 = d0 * rstd * gv.x + bb.x, n1 = d1 * rstd * gv.y + bb.y;
        const float n2 = d2 * rstd * gv.z + bb.z, n3 = d3 * rstd * gv.w + bb.w;
        uint2 pk; pk.x = packbf(n0, n1); pk.y = packbf(n2, n3);
        *(uint2*)&Ap[m * 68 + 2 * lane] = pk;
    }
    __syncthreads();

    // ---- stage swizzled W1 (overwrites w_dw region): linear uint4 copy ----
    {
        const uint4* src = (const uint4*)w1p_g;
        uint4* dst = (uint4*)Bsw;
        for (int t = tid; t < 8192; t += 256) dst[t] = src[t];
    }
    __syncthreads();

    // ---- WMMA: wave wv -> m-tile wv, 2 n-tiles per pass, K=128 in 4 steps ----
    const int mbase = wv * 16;
    const int mrow  = lane & 15;
    const int half  = lane >> 4;
    const int ccol  = lane & 15;
    for (int np = 0; np < 16; ++np) {
        const int nt0 = np * 2, nt1 = np * 2 + 1;
        v8f acc0 = {0.f, 0.f, 0.f, 0.f, 0.f, 0.f, 0.f, 0.f};
        v8f acc1 = {0.f, 0.f, 0.f, 0.f, 0.f, 0.f, 0.f, 0.f};
#pragma unroll
        for (int ks = 0; ks < 4; ++ks) {
            AV a, b0, b1;
            const unsigned int* arow = &Ap[(mbase + mrow) * 68 + ks * 16 + half * 4];
            a.q[0] = *(const uint4*)arow;
            a.q[1] = *(const uint4*)(arow + 8);
            const int g0 = ((ks * 512 + nt0 * 16 + ccol) * 2 + half) * 4;
            const int g1 = ((ks * 512 + nt1 * 16 + ccol) * 2 + half) * 4;
            b0.q[0] = *(const uint4*)&Bsw[g0];
            b0.q[1] = *(const uint4*)&Bsw[16384 + g0];
            b1.q[0] = *(const uint4*)&Bsw[g1];
            b1.q[1] = *(const uint4*)&Bsw[16384 + g1];
            acc0 = __builtin_amdgcn_wmma_f32_16x16x32_bf16(false, a.v, false, b0.v,
                                                           (short)0, acc0, false, false);
            acc1 = __builtin_amdgcn_wmma_f32_16x16x32_bf16(false, a.v, false, b1.v,
                                                           (short)0, acc1, false, false);
        }
        epilogue1(acc0, nt0, blockRow, mbase, half, ccol, b1s, Hp, sumsq);
        epilogue1(acc1, nt1, blockRow, mbase, half, ccol, b1s, Hp, sumsq);
    }
}

// ---------------------------------------------------------------------------
// K2a: gx = sqrt(sumsq); nx = gx/(mean(gx)+1e-6); s = grn_g*nx + 1
// ---------------------------------------------------------------------------
__global__ void k_grn_stats(const float* __restrict__ sumsq, const float* __restrict__ grn_g,
                            float* __restrict__ sbuf) {
    __shared__ float red[HIDC];
    const int t = threadIdx.x;               // 512 threads
    const float gx = sqrtf(sumsq[t]);
    red[t] = gx;
    __syncthreads();
    for (int o = 256; o >= 1; o >>= 1) {
        if (t < o) red[t] += red[t + o];
        __syncthreads();
    }
    const float mean = red[0] * (1.0f / 512.0f);
    sbuf[t] = grn_g[t] * (gx / (mean + 1e-6f)) + 1.0f;
}

// ---------------------------------------------------------------------------
// K2b: w2s[k][n] = s[k]*w2[k][n], bf16 K-pairs in swizzled-B layout (N=128,S=16).
// ---------------------------------------------------------------------------
__global__ void k_make_w2s(const float* __restrict__ w2, const float* __restrict__ sbuf,
                           unsigned int* __restrict__ w2sp) {
    int t = blockIdx.x * blockDim.x + threadIdx.x;   // 32768
    if (t < 32768) {
        int p  = t >> 14;
        int rr = t & 16383;
        int g  = rr >> 2, c = rr & 3;
        int half = g & 1, rest = g >> 1;
        int col = rest & 127, ks = rest >> 7;
        int k2  = ks * 16 + p * 8 + half * 4 + c;
        w2sp[t] = packbf(sbuf[2 * k2] * w2[(2 * k2) * DIMC + col],
                         sbuf[2 * k2 + 1] * w2[(2 * k2 + 1) * DIMC + col]);
    }
}

// ---------------------------------------------------------------------------
// K3: GEMM2 (H_bf16 @ scaled-W2, 4 concurrent n-tiles) + folded bias + residual.
// ---------------------------------------------------------------------------
__global__ __launch_bounds__(256) void k_layer_back(
        const unsigned int* __restrict__ Hp, const unsigned int* __restrict__ w2sp,
        const float* __restrict__ cb, const float* __restrict__ xres,
        float* __restrict__ out) {
    __shared__ unsigned int Bsw[32768];     // scaled W2, swizzled (2 planes)
    __shared__ unsigned int Ap[64 * 260];   // H tile bf16 pairs, padded row stride
    const int tid  = threadIdx.x;
    const int lane = tid & 31, wv = tid >> 5;
    const int blockRow = blockIdx.x * 64;

    {   // stage A: 64 rows x 256 dwords; stage B: linear copy
        const uint4* src = (const uint4*)(Hp + (long)blockRow * 256);
        for (int t = tid; t < 64 * 64; t += 256) {
            int r = t >> 6, c4 = t & 63;
            *(uint4*)&Ap[r * 260 + c4 * 4] = src[r * 64 + c4];
        }
        const uint4* bs = (const uint4*)w2sp;
        uint4* dst = (uint4*)Bsw;
        for (int t = tid; t < 8192; t += 256) dst[t] = bs[t];
    }
    __syncthreads();

    const int mt   = wv >> 1;
    const int mrow = lane & 15, half = lane >> 4, ccol = lane & 15;
    const int ntb  = (wv & 1) * 4;

    v8f acc[4];
#pragma unroll
    for (int t = 0; t < 4; ++t) acc[t] = (v8f){0.f, 0.f, 0.f, 0.f, 0.f, 0.f, 0.f, 0.f};

    for (int ks = 0; ks < 16; ++ks) {
        AV a;
        const unsigned int* arow = &Ap[(mt * 16 + mrow) * 260 + ks * 16 + half * 4];
        a.q[0] = *(const uint4*)arow;
        a.q[1] = *(const uint4*)(arow + 8);
#pragma unroll
        for (int t = 0; t < 4; ++t) {
            AV bm;
            const int g = ((ks * 128 + (ntb + t) * 16 + ccol) * 2 + half) * 4;
            bm.q[0] = *(const uint4*)&Bsw[g];
            bm.q[1] = *(const uint4*)&Bsw[16384 + g];
            acc[t] = __builtin_amdgcn_wmma_f32_16x16x32_bf16(false, a.v, false, bm.v,
                                                             (short)0, acc[t], false, false);
        }
    }
#pragma unroll
    for (int t = 0; t < 4; ++t) {
        const int col  = (ntb + t) * 16 + ccol;
        const float bias = cb[col];
#pragma unroll
        for (int r = 0; r < 8; ++r) {
            const int row = blockRow + mt * 16 + r + 8 * half;
            out[(long)row * DIMC + col] = acc[t][r] + bias + xres[(long)row * DIMC + col];
        }
    }
}

// ---------------------------------------------------------------------------
extern "C" void kernel_launch(void* const* d_in, const int* in_sizes, int n_in,
                              void* d_out, int out_size, void* d_ws, size_t ws_size,
                              hipStream_t stream) {
    const float* feats = (const float*)d_in[0];
    const float* w_dw  = (const float*)d_in[1];
    const float* b_dw  = (const float*)d_in[2];
    const float* ln_g  = (const float*)d_in[3];
    const float* ln_b  = (const float*)d_in[4];
    const float* w1    = (const float*)d_in[5];
    const float* b1    = (const float*)d_in[6];
    const float* grn_g = (const float*)d_in[7];
    const float* grn_b = (const float*)d_in[8];
    const float* w2    = (const float*)d_in[9];
    const float* b2    = (const float*)d_in[10];
    const int*   nbr   = (const int*)d_in[11];
    float* out = (float*)d_out;

    char* ws = (char*)d_ws;
    size_t off = 0;
    unsigned int* w1p = (unsigned int*)(ws + off); off += (size_t)DEPTHL * 32768 * 4;
    float* cb    = (float*)(ws + off); off += (size_t)DEPTHL * DIMC * 4;
    float* sumsq = (float*)(ws + off); off += (size_t)DEPTHL * HIDC * 4;
    float* sbuf  = (float*)(ws + off); off += (size_t)HIDC * 4;
    off = (off + 255) & ~(size_t)255;
    unsigned int* w2sp = (unsigned int*)(ws + off); off += (size_t)32768 * 4;
    off = (off + 255) & ~(size_t)255;
    unsigned int* Hp = (unsigned int*)(ws + off); off += (size_t)NROWS * 256 * 4;
    off = (off + 255) & ~(size_t)255;
    float* X0 = (float*)(ws + off);

    k_prep<<<256, 256, 0, stream>>>(w1, grn_b, w2, b2, w1p, cb, sumsq);

    for (int i = 0; i < DEPTHL; ++i) {
        const float* x = (i == 0) ? feats : X0;
        float*       y = (i == 0) ? X0 : out;
        k_layer_front<<<NROWS / 128, 256, 0, stream>>>(
            x, nbr, w_dw + i * KK * DIMC, b_dw + i * DIMC, ln_g + i * DIMC, ln_b + i * DIMC,
            w1p + i * 32768, b1 + i * HIDC, Hp, sumsq + i * HIDC);
        k_grn_stats<<<1, 512, 0, stream>>>(sumsq + i * HIDC, grn_g + i * HIDC, sbuf);
        k_make_w2s<<<128, 256, 0, stream>>>(w2 + i * HIDC * DIMC, sbuf, w2sp);
        k_layer_back<<<NROWS / 64, 256, 0, stream>>>(Hp, w2sp, cb + i * DIMC, x, y);
    }
}
